// AttentionWeighting_43722767073326
// MI455X (gfx1250) — matrix operand order repaired and checked by reading, not compiled, not run on hardware
//
#include <hip/hip_runtime.h>
#include <hip/hip_bf16.h>
#include <math.h>

#define N_PTS   150000
#define C_IN    256
#define C_OUT   64
#define KTAPS   9
#define EPS_F   1e-5f
#define TILES   (N_PTS / 16)        // 9375 exactly, no tail
#define KB_STEPS (C_IN / 32)        // 8 K-blocks of 32
#define NT      (C_OUT / 16)        // 4 N-tiles of 16
#define WAVES_PER_BLK 4

typedef __attribute__((ext_vector_type(16))) __bf16 v16bf;
typedef __attribute__((ext_vector_type(8)))  float  v8f;

// ---------------------------------------------------------------------------
// Prologue: split W1 (fp32 [9][256][64]) into bf16 hi/lo fragments laid out
// as WMMA B-operand fragments in the same k order as the A side:
//   fragment f = (tap*8 + kb)*4 + nt, 512 bf16 elems per fragment,
//   lane L holds column N = nt*16 + (L&15), K = kb*32 + (L<16?0:16) + e.
// ---------------------------------------------------------------------------
__global__ void build_w1_frags(const float* __restrict__ W1,
                               __bf16* __restrict__ fragHi,
                               __bf16* __restrict__ fragLo) {
    int f    = blockIdx.x;          // 0..287
    int lane = threadIdx.x;         // 0..31
    int nt   = f & 3;
    int kb   = (f >> 2) & 7;
    int tap  = f >> 5;
    int kbase = kb * 32 + ((lane & 16) ? 16 : 0);
    int n     = nt * 16 + (lane & 15);
    size_t ob = (size_t)f * 512 + (size_t)lane * 16;
#pragma unroll
    for (int e = 0; e < 16; ++e) {
        float v  = W1[((size_t)tap * C_IN + (size_t)(kbase + e)) * C_OUT + n];
        __bf16 h = (__bf16)v;
        fragHi[ob + e] = h;
        fragLo[ob + e] = (__bf16)(v - (float)h);
    }
}

__global__ void zero_stats(float* stats) {
    stats[threadIdx.x] = 0.0f;      // sum[64], sumsq[64]
}

// ---------------------------------------------------------------------------
// Helper: build bf16 hi/lo A-fragments for one K-block from a gathered row.
// ---------------------------------------------------------------------------
__device__ __forceinline__ void load_a_frag(const float* rp, bool valid,
                                            int kb, int koff,
                                            v16bf& ahi, v16bf& alo) {
    float xs[16];
    if (valid) {
        const float4* p0 = (const float4*)(rp + kb * 32 + koff);
        const float4* p1 = (const float4*)(rp + kb * 32 + 8 + koff);
        float4 f0 = p0[0], f1 = p0[1];
        float4 f2 = p1[0], f3 = p1[1];
        xs[0]=f0.x;  xs[1]=f0.y;  xs[2]=f0.z;  xs[3]=f0.w;
        xs[4]=f1.x;  xs[5]=f1.y;  xs[6]=f1.z;  xs[7]=f1.w;
        xs[8]=f2.x;  xs[9]=f2.y;  xs[10]=f2.z; xs[11]=f2.w;
        xs[12]=f3.x; xs[13]=f3.y; xs[14]=f3.z; xs[15]=f3.w;
    } else {
#pragma unroll
        for (int e = 0; e < 16; ++e) xs[e] = 0.0f;
    }
#pragma unroll
    for (int e = 0; e < 16; ++e) {
        __bf16 h = (__bf16)xs[e];
        ahi[e] = h;
        alo[e] = (__bf16)(xs[e] - (float)h);
    }
}

// ---------------------------------------------------------------------------
// Helper: one K=32 step of the 3xBF16-split GEMM across all 4 N-tiles,
// ordered so dependent WMMAs on the same accumulator are 4 issues apart.
// ---------------------------------------------------------------------------
__device__ __forceinline__ void wmma_step(const v16bf& ahi, const v16bf& alo,
                                          const __bf16* __restrict__ fragHi,
                                          const __bf16* __restrict__ fragLo,
                                          size_t fb, v8f cacc[NT]) {
    v16bf bh[NT], bl[NT];
#pragma unroll
    for (int nt = 0; nt < NT; ++nt) {
        bh[nt] = *(const v16bf*)(fragHi + fb + (size_t)nt * 512);
        bl[nt] = *(const v16bf*)(fragLo + fb + (size_t)nt * 512);
    }
#pragma unroll
    for (int nt = 0; nt < NT; ++nt)
        cacc[nt] = __builtin_amdgcn_wmma_f32_16x16x32_bf16(
            false, ahi, false, bh[nt], (short)0, cacc[nt], false, false);
#pragma unroll
    for (int nt = 0; nt < NT; ++nt)
        cacc[nt] = __builtin_amdgcn_wmma_f32_16x16x32_bf16(
            false, alo, false, bh[nt], (short)0, cacc[nt], false, false);
#pragma unroll
    for (int nt = 0; nt < NT; ++nt)
        cacc[nt] = __builtin_amdgcn_wmma_f32_16x16x32_bf16(
            false, ahi, false, bl[nt], (short)0, cacc[nt], false, false);
}

// ---------------------------------------------------------------------------
// Main kernel: one wave per 16-row tile. Gather -> 3xBF16-split WMMA GEMM
// over 9 taps x 8 K-blocks x 4 N-tiles, fp32 accumulate. Writes acc and
// channel sum / sum-of-squares partials for the batch-norm.
// ---------------------------------------------------------------------------
__global__ void __launch_bounds__(32 * WAVES_PER_BLK)
spconv_wmma(const float* __restrict__ feats,
            const long long* __restrict__ in_map,
            const __bf16* __restrict__ fragHi,
            const __bf16* __restrict__ fragLo,
            float* __restrict__ acc,
            float* __restrict__ stat_sum,
            float* __restrict__ stat_sq) {
    __shared__ float s_sum[C_OUT];
    __shared__ float s_sq[C_OUT];
    int tid = threadIdx.x;
    if (tid < C_OUT) { s_sum[tid] = 0.0f; s_sq[tid] = 0.0f; }
    __syncthreads();

    int wave = tid >> 5;
    int lane = tid & 31;
    int tile = blockIdx.x * WAVES_PER_BLK + wave;

    if (tile < TILES) {
        int m    = lane & 15;                 // output row within tile
        int koff = (lane & 16) ? 16 : 0;      // K sub-offset per A layout
        int base = tile * 16;

        v8f cacc[NT] = {};                    // 4 x (16x16 f32) accumulators

        for (int tap = 0; tap < KTAPS; ++tap) {
            long long rowIdx = in_map[(size_t)tap * N_PTS + base + m];
            bool valid = (rowIdx < (long long)N_PTS);   // sentinel == N
            const float* rp = feats + (size_t)(valid ? rowIdx : 0) * C_IN;

            // Warm next tap's scattered row (global_prefetch_b8).
            if (tap + 1 < KTAPS) {
                long long nIdx = in_map[(size_t)(tap + 1) * N_PTS + base + m];
                if (nIdx < (long long)N_PTS) {
                    const float* np = feats + (size_t)nIdx * C_IN + koff;
                    __builtin_prefetch(np, 0, 3);
                    __builtin_prefetch(np + 128, 0, 3);
                }
            }

#pragma unroll 2
            for (int kb = 0; kb < KB_STEPS; ++kb) {
                v16bf ahi, alo;
                load_a_frag(rp, valid, kb, koff, ahi, alo);
                size_t fb = (size_t)((tap * KB_STEPS + kb) * NT) * 512
                          + (size_t)lane * 16;
                wmma_step(ahi, alo, fragHi, fragLo, fb, cacc);
            }
        }

        // Store acc + per-channel partial stats.
        // C/D layout: VGPR r -> row (lane<16 ? r : 8+r), col = lane&15.
        int colb  = lane & 15;
        int rbase = (lane & 16) ? 8 : 0;
#pragma unroll
        for (int nt = 0; nt < NT; ++nt) {
            int col = nt * 16 + colb;
            float psum = 0.0f, psq = 0.0f;
#pragma unroll
            for (int r = 0; r < 8; ++r) {
                float v = cacc[nt][r];
                acc[(size_t)(base + rbase + r) * C_OUT + col] = v;
                psum += v;
                psq  += v * v;
            }
            atomicAdd(&s_sum[col], psum);
            atomicAdd(&s_sq[col],  psq);
        }
    }

    __syncthreads();
    if (tid < C_OUT) {
        atomicAdd(&stat_sum[tid], s_sum[tid]);
        atomicAdd(&stat_sq[tid],  s_sq[tid]);
    }
}

// ---------------------------------------------------------------------------
// Finalize batch-norm statistics.
// ---------------------------------------------------------------------------
__global__ void bn_finalize(const float* __restrict__ stat_sum,
                            const float* __restrict__ stat_sq,
                            float* __restrict__ mean,
                            float* __restrict__ inv) {
    int c = threadIdx.x;            // 64
    float m = stat_sum[c] * (1.0f / (float)N_PTS);
    float v = stat_sq[c]  * (1.0f / (float)N_PTS) - m * m;
    mean[c] = m;
    inv[c]  = rsqrtf(v + EPS_F);
}

// ---------------------------------------------------------------------------
// Epilogue: normalize, ReLU, 64->1 projection, sigmoid. One wave per row,
// each lane covers channels {l, l+32}, wave32 shuffle reduction.
// ---------------------------------------------------------------------------
__global__ void __launch_bounds__(256)
epilogue(const float* __restrict__ acc,
         const float* __restrict__ gamma,
         const float* __restrict__ beta,
         const float* __restrict__ W2,
         const float* __restrict__ mean,
         const float* __restrict__ inv,
         float* __restrict__ out) {
    int lane = threadIdx.x & 31;
    int warp = threadIdx.x >> 5;
    int row  = blockIdx.x * 8 + warp;
    if (row >= N_PTS) return;

    float s = 0.0f;
#pragma unroll
    for (int h = 0; h < 2; ++h) {
        int c   = lane + 32 * h;
        float x = acc[(size_t)row * C_OUT + c];
        float y = gamma[c] * (x - mean[c]) * inv[c] + beta[c];
        y = fmaxf(y, 0.0f);
        s += y * W2[c];
    }
#pragma unroll
    for (int off = 16; off > 0; off >>= 1)
        s += __shfl_down(s, off, 32);
    if (lane == 0)
        out[row] = 1.0f / (1.0f + __expf(-s));
}

// ---------------------------------------------------------------------------
extern "C" void kernel_launch(void* const* d_in, const int* in_sizes, int n_in,
                              void* d_out, int out_size, void* d_ws, size_t ws_size,
                              hipStream_t stream) {
    const float*     feats  = (const float*)d_in[0];
    const float*     W1     = (const float*)d_in[1];
    const float*     gamma  = (const float*)d_in[2];
    const float*     beta   = (const float*)d_in[3];
    const float*     W2     = (const float*)d_in[4];
    const long long* in_map = (const long long*)d_in[5];
    // d_in[6] (out_map) is redundant: out_map[k][j]==j exactly when
    // in_map[k][j] != N, which the kernel already checks.
    float* out = (float*)d_out;

    char*  ws        = (char*)d_ws;
    size_t accBytes  = (size_t)N_PTS * C_OUT * sizeof(float);          // 38.4 MB
    size_t fragElems = (size_t)KTAPS * KB_STEPS * NT * 512;            // 147456

    float*  acc     = (float*)ws;
    __bf16* fragHi  = (__bf16*)(ws + accBytes);
    __bf16* fragLo  = fragHi + fragElems;
    float*  stats   = (float*)(ws + accBytes + 2 * fragElems * sizeof(__bf16));
    float*  statSum = stats;        // [64]
    float*  statSq  = stats + 64;   // [64]
    float*  bnMean  = stats + 128;  // [64]
    float*  bnInv   = stats + 192;  // [64]

    build_w1_frags<<<KTAPS * KB_STEPS * NT, 32, 0, stream>>>(W1, fragHi, fragLo);
    zero_stats<<<1, 128, 0, stream>>>(stats);

    int nblk = (TILES + WAVES_PER_BLK - 1) / WAVES_PER_BLK;
    spconv_wmma<<<nblk, 32 * WAVES_PER_BLK, 0, stream>>>(
        feats, in_map, fragHi, fragLo, acc, statSum, statSq);

    bn_finalize<<<1, 64, 0, stream>>>(statSum, statSq, bnMean, bnInv);

    epilogue<<<N_PTS / 8, 256, 0, stream>>>(
        acc, gamma, beta, W2, bnMean, bnInv, out);
}
